// SlidingWindowAttention_56573309223857
// MI455X (gfx1250) — matrix operand order repaired and checked by reading, not compiled, not run on hardware
//
#include <hip/hip_runtime.h>

typedef __attribute__((ext_vector_type(16))) _Float16 v16h;
typedef __attribute__((ext_vector_type(8)))  float    v8f;
typedef __attribute__((ext_vector_type(2)))  __fp16   fp16x2;

#define S_LEN   4096
#define HEADS   16
#define DIM     64
#define HALF_W  256
#define BQ      64
#define NEG_BIG (-1.0e30f)

union PkH2 { fp16x2 v; unsigned u; _Float16 h[2]; };

// load 2x16B from LDS into one v16h fragment (halves 0-7 from p0, 8-15 from p1)
__device__ __forceinline__ v16h ld_frag_split(const _Float16* p0, const _Float16* p1) {
    union { uint4 u[2]; v16h v; } t;
    t.u[0] = *(const uint4*)p0;
    t.u[1] = *(const uint4*)p1;
    return t.v;
}

__global__ __launch_bounds__(128)
void swa_fwd_kernel(const float* __restrict__ Qg0, const float* __restrict__ Kg0,
                    const float* __restrict__ Vg0, float* __restrict__ Og) {
    __shared__ _Float16 Qs[BQ * DIM];       // 8 KB, query tile (f16, row-major)
    __shared__ _Float16 Ks[64 * DIM];       // 8 KB, key tile   (f16, row-major)
    __shared__ _Float16 Vt[DIM * 64];       // 8 KB, value tile (f16, [d][perm-key])
    __shared__ _Float16 Ps[4 * 16 * 64];    // 8 KB, per-wave P scratch 16x64 (perm cols)

    const int tid  = threadIdx.x;
    const int lid  = tid & 31;
    const int wv   = tid >> 5;      // wave id 0..3
    const int hi16 = lid >> 4;      // half-wave: 0 = lanes 0-15, 1 = lanes 16-31
    const int l16  = lid & 15;

    const int h  = blockIdx.y;
    const int q0 = blockIdx.x * BQ;

    const float* Qg = Qg0 + ((size_t)h * S_LEN + q0) * DIM;
    const float* Kg = Kg0 + (size_t)h * S_LEN * DIM;
    const float* Vg = Vg0 + (size_t)h * S_LEN * DIM;

    // fold softmax scale (1/sqrt(64)) and log2(e) into Q: p = exp2(s - m)
    const float qscale = 0.125f * 1.44269504088896340736f;

    // ---- stage Q tile (f32 -> f16, pre-scaled, packed cvt) -----------------
    #pragma unroll
    for (int j = 0; j < 8; ++j) {
        int i   = tid + j * 128;          // 0..1023 float4 slots
        int row = i >> 4;
        int c4  = (i & 15) << 2;
        float4 f = *(const float4*)(Qg + row * DIM + c4);
        PkH2 a, b;
        a.v = __builtin_amdgcn_cvt_pkrtz(f.x * qscale, f.y * qscale);
        b.v = __builtin_amdgcn_cvt_pkrtz(f.z * qscale, f.w * qscale);
        uint2 pk; pk.x = a.u; pk.y = b.u;
        *(uint2*)(&Qs[row * DIM + c4]) = pk;
    }
    __syncthreads();

    // ---- Q A-fragments: 16 rows x 32 (two d-chunks), ISA 16-bit A layout ---
    v16h qf[2];
    #pragma unroll
    for (int c = 0; c < 2; ++c) {
        const _Float16* b0 = &Qs[(wv * 16 + l16) * DIM + c * 32 + hi16 * 8];
        qf[c] = ld_frag_split(b0, b0 + 16);
    }

    v8f acc[4];
    float m_r[8], l_r[8];
    #pragma unroll
    for (int r = 0; r < 8; ++r) {
        m_r[r] = NEG_BIG; l_r[r] = 0.0f;
        #pragma unroll
        for (int t = 0; t < 4; ++t) acc[t][r] = 0.0f;
    }

    int ktS = q0 - HALF_W;      if (ktS < 0)      ktS = 0;
    int ktE = q0 + BQ + HALF_W; if (ktE > S_LEN)  ktE = S_LEN;

    const int qa   = q0 + wv * 16;            // wave's first query row
    const int qRow = qa + 8 * hi16;           // + r gives this lane's row

    for (int kt = ktS; kt < ktE; kt += 64) {
        __syncthreads();   // previous tile fully consumed before overwrite
        // ---- stage K tile (row-major) and V tile (transposed + permuted) --
        #pragma unroll
        for (int j = 0; j < 8; ++j) {
            int i   = tid + j * 128;
            int row = i >> 4;
            int c4  = (i & 15) << 2;
            float4 f = *(const float4*)(Kg + (size_t)(kt + row) * DIM + c4);
            PkH2 a, b;
            a.v = __builtin_amdgcn_cvt_pkrtz(f.x, f.y);
            b.v = __builtin_amdgcn_cvt_pkrtz(f.z, f.w);
            uint2 pk; pk.x = a.u; pk.y = b.u;
            *(uint2*)(&Ks[row * DIM + c4]) = pk;

            // permuted key column: storage col c holds key (c&3)*16 + (c>>2)
            int cp = ((row & 15) << 2) + (row >> 4);
            float4 g = *(const float4*)(Vg + (size_t)(kt + row) * DIM + c4);
            PkH2 va, vb;
            va.v = __builtin_amdgcn_cvt_pkrtz(g.x, g.y);
            vb.v = __builtin_amdgcn_cvt_pkrtz(g.z, g.w);
            Vt[(c4 + 0) * 64 + cp] = va.h[0];
            Vt[(c4 + 1) * 64 + cp] = va.h[1];
            Vt[(c4 + 2) * 64 + cp] = vb.h[0];
            Vt[(c4 + 3) * 64 + cp] = vb.h[1];
        }
        __syncthreads();

        // prefetch next K/V tile: 128 threads x 128B lines covers 16KB each
        if (kt + 64 < ktE) {
            __builtin_prefetch(Kg + (size_t)(kt + 64) * DIM + tid * 32, 0, 3);
            __builtin_prefetch(Vg + (size_t)(kt + 64) * DIM + tid * 32, 0, 3);
        }

        // wave-uniform guards (scalar branch; EXEC stays all-ones for WMMA)
        const bool active   = (kt <= qa + 15 + HALF_W) && (kt + 63 >= qa - HALF_W);
        const bool needMask = (kt < qa - 241) || (kt > qa + 193);
        if (active) {
            // ---- S = Q * K^T : 4 subtiles x 2 K=32 WMMAs ------------------
            v8f sc[4];
            #pragma unroll
            for (int n = 0; n < 4; ++n) {
                v8f s0 = {};
                #pragma unroll
                for (int c = 0; c < 2; ++c) {
                    const _Float16* b0 = &Ks[(n * 16 + l16) * DIM + c * 32 + hi16 * 8];
                    v16h kf = ld_frag_split(b0, b0 + 16);
                    s0 = __builtin_amdgcn_wmma_f32_16x16x32_f16(
                             false, qf[c], false, kf, (short)0, s0, false, false);
                }
                sc[n] = s0;
            }

            // ---- band mask only on boundary tiles -------------------------
            if (needMask) {
                #pragma unroll
                for (int n = 0; n < 4; ++n) {
                    const int kcol = kt + n * 16 + l16;
                    #pragma unroll
                    for (int r = 0; r < 8; ++r) {
                        int dq = qRow + r - kcol; if (dq < 0) dq = -dq;
                        sc[n][r] = (dq <= HALF_W) ? sc[n][r] : NEG_BIG;
                    }
                }
            }

            // ---- batched online softmax: one reduction per row per tile ---
            #pragma unroll
            for (int r = 0; r < 8; ++r) {
                float x = fmaxf(fmaxf(sc[0][r], sc[1][r]), fmaxf(sc[2][r], sc[3][r]));
                x = fmaxf(x, __shfl_xor(x, 1, 32));
                x = fmaxf(x, __shfl_xor(x, 2, 32));
                x = fmaxf(x, __shfl_xor(x, 4, 32));
                x = fmaxf(x, __shfl_xor(x, 8, 32));

                float mp    = fmaxf(m_r[r], x);
                float alpha = __builtin_amdgcn_exp2f(m_r[r] - mp);

                float p0 = __builtin_amdgcn_exp2f(sc[0][r] - mp);
                float p1 = __builtin_amdgcn_exp2f(sc[1][r] - mp);
                float p2 = __builtin_amdgcn_exp2f(sc[2][r] - mp);
                float p3 = __builtin_amdgcn_exp2f(sc[3][r] - mp);

                // packed store: this lane's 4 probs land in contiguous
                // permuted columns c = l16*4 + n  (one ds_store_b64 per row)
                PkH2 pa, pb2;
                pa.v  = __builtin_amdgcn_cvt_pkrtz(p0, p1);
                pb2.v = __builtin_amdgcn_cvt_pkrtz(p2, p3);
                uint2 pk; pk.x = pa.u; pk.y = pb2.u;
                *(uint2*)(&Ps[wv * 1024 + (r + 8 * hi16) * 64 + (l16 << 2)]) = pk;

                float ps = (p0 + p1) + (p2 + p3);
                ps += __shfl_xor(ps, 1, 32);
                ps += __shfl_xor(ps, 2, 32);
                ps += __shfl_xor(ps, 4, 32);
                ps += __shfl_xor(ps, 8, 32);

                l_r[r] = l_r[r] * alpha + ps;
                m_r[r] = mp;
                #pragma unroll
                for (int t = 0; t < 4; ++t) acc[t][r] *= alpha;
            }

            // same-wave LDS store->load: DS pipe is in-order, wait for data
            asm volatile("s_wait_dscnt 0x0" ::: "memory");

            // ---- O += P * V : A = P (16 x 32 perm-keys per chunk),
            //      B = Vt chunk (32 perm-keys x 16 d) — same K-permutation ---
            #pragma unroll
            for (int chunk = 0; chunk < 2; ++chunk) {
                const _Float16* pb = &Ps[wv * 1024 + l16 * 64 + chunk * 32 + hi16 * 8];
                v16h pf = ld_frag_split(pb, pb + 16);
                #pragma unroll
                for (int t = 0; t < 4; ++t) {
                    const _Float16* vb = &Vt[(t * 16 + l16) * 64 + chunk * 32 + hi16 * 8];
                    v16h vf = ld_frag_split(vb, vb + 16);
                    acc[t] = __builtin_amdgcn_wmma_f32_16x16x32_f16(
                                 false, pf, false, vf, (short)0, acc[t], false, false);
                }
            }
        }
    }

    // ---- epilogue: O = acc * (1/l) ----------------------------------------
    #pragma unroll
    for (int r = 0; r < 8; ++r) {
        const float inv = __builtin_amdgcn_rcpf(l_r[r]);
        const int qg = qRow + r;
        #pragma unroll
        for (int t = 0; t < 4; ++t) {
            Og[((size_t)h * S_LEN + qg) * DIM + t * 16 + l16] = acc[t][r] * inv;
        }
    }
}

extern "C" void kernel_launch(void* const* d_in, const int* in_sizes, int n_in,
                              void* d_out, int out_size, void* d_ws, size_t ws_size,
                              hipStream_t stream) {
    const float* q = (const float*)d_in[0];
    const float* k = (const float*)d_in[1];
    const float* v = (const float*)d_in[2];
    float* out = (float*)d_out;
    (void)in_sizes; (void)n_in; (void)out_size; (void)d_ws; (void)ws_size;

    dim3 grid(S_LEN / BQ, HEADS, 1);   // (64, 16)
    dim3 block(128, 1, 1);             // 4 waves of 32
    swa_fwd_kernel<<<grid, block, 0, stream>>>(q, k, v, out);
}
